// GraphNetBlock_10393820856378
// MI455X (gfx1250) — matrix operand order repaired and checked by reading, not compiled, not run on hardware
//
#include <hip/hip_runtime.h>

#define NN 50000
#define EE 800000
#define ND 128
#define ED 128
#define HD 256

typedef __attribute__((ext_vector_type(16))) _Float16 v16h;
typedef __attribute__((ext_vector_type(8)))  float    v8f;

// ---------------------------------------------------------------------------
// Fragment helpers (CDNA5 WMMA 16x16x32 f16, wave32; layouts per ISA 7.12.2)
// ---------------------------------------------------------------------------

// A-matrix 16x32 f16 from LDS (row-major, stride lda halfs).
// lane 0-15: M=lane, K in {0..7}U{16..23}; lane 16-31: M=lane-16, K in {8..15}U{24..31}
__device__ __forceinline__ v16h lds_a_frag(const _Float16* A, int lda) {
  const int lane = threadIdx.x & 31;
  const int m = lane & 15, h = lane >> 4;
  const _Float16* row = A + m * lda;
  v16h a;
#pragma unroll
  for (int p = 0; p < 8; ++p) {
    const int k = ((p & 4) ? 16 : 0) + h * 8 + (p & 3) * 2;
    a[2 * p]     = row[k];
    a[2 * p + 1] = row[k + 1];
  }
  return a;
}

// B-matrix 32x16 f16 from pre-swizzled global weights: one contiguous
// 32-byte load per lane. Fragment (kt, nt) of a K x N weight matrix.
__device__ __forceinline__ v16h ldg_b_frag(const _Float16* __restrict__ Wsw,
                                           int Kt, int kt, int nt) {
  const int lane = threadIdx.x & 31;
  return *(const v16h*)(Wsw + ((size_t)((nt * Kt + kt) * 32 + lane) << 4));
}

__device__ __forceinline__ float silu_f(float v) { return v / (1.0f + __expf(-v)); }

// ---------------------------------------------------------------------------
// Prep kernels
// ---------------------------------------------------------------------------

__global__ void zero_f32_kernel(float* __restrict__ p, int n) {
  int i = blockIdx.x * blockDim.x + threadIdx.x;
  if (i < n) p[i] = 0.0f;
}

// Convert fp32 weight W[K][Nout] (row-major) into f16 B-fragment-swizzled
// layout: frag (kt, nt), 32 lanes x 16 halfs contiguous per lane.
// b[j] of lane L corresponds to B[kt*32 + (j>=8?16:0) + (L>=16?8:0) + (j&7)][nt*16 + (L&15)].
__global__ void swizzle_w_kernel(const float* __restrict__ W,
                                 _Float16* __restrict__ out, int K, int Nout) {
  const int t = blockIdx.x * blockDim.x + threadIdx.x;
  const int nfrag = (K / 32) * (Nout / 16);
  if (t >= nfrag * 32) return;
  const int lane = t & 31, f = t >> 5;
  const int kt = f % (K / 32), nt = f / (K / 32);
  const int n = lane & 15, h = lane >> 4;
  _Float16* dst = out + ((size_t)t << 4);
#pragma unroll 4
  for (int j = 0; j < 16; ++j) {
    const int k = kt * 32 + ((j & 8) ? 16 : 0) + h * 8 + (j & 7);
    dst[j] = (_Float16)W[(size_t)k * Nout + nt * 16 + n];
  }
}

// ---------------------------------------------------------------------------
// Fused edge-update kernel:
//   Xe=[64,384] concat gather -> H=silu(Xe@W1+b1) [64,256]
//   -> O=H@W2+Xe@Wr+b2+br [64,128] -> LayerNorm -> new_edge + atomic scatter
// ---------------------------------------------------------------------------
__global__ void __launch_bounds__(256)
edge_kernel(const float* __restrict__ x, const float* __restrict__ eattr,
            const long long* __restrict__ eidx,
            const _Float16* __restrict__ w1s, const float* __restrict__ b1,
            const _Float16* __restrict__ w2s, const float* __restrict__ b2,
            const _Float16* __restrict__ wrs, const float* __restrict__ br,
            const float* __restrict__ gamma, const float* __restrict__ beta,
            float* __restrict__ out_edge, float* __restrict__ agg) {
  extern __shared__ char smem[];
  _Float16* sX = (_Float16*)smem;                              // 64 x 384 f16
  _Float16* sH = (_Float16*)(smem + 64 * 384 * 2);             // 64 x 256 f16
  float*    sO = (float*)(smem + 64 * 384 * 2 + 64 * 256 * 2); // 64 x 132 f32

  const int tid = threadIdx.x;
  const int wv = tid >> 5;
  const int lane = tid & 31;
  const int n16 = lane & 15, hlf = lane >> 4;
  const int e0 = blockIdx.x * 64;

  // Stage A: gather [edge_attr | x[src] | x[dst]] -> f16 LDS
  for (int i = tid; i < 64 * 384; i += 256) {
    const int r = i / 384, c = i - r * 384;
    float v;
    if (c < ED) {
      v = eattr[(size_t)(e0 + r) * ED + c];
    } else if (c < ED + ND) {
      const int s = (int)eidx[e0 + r];
      v = x[(size_t)s * ND + (c - ED)];
    } else {
      const int d = (int)eidx[EE + e0 + r];
      v = x[(size_t)d * ND + (c - ED - ND)];
    }
    sX[i] = (_Float16)v;
  }
  __syncthreads();

  // Stage B: H = silu(Xe @ W1 + b1), 64x256, K=384 (12 k-tiles)
  for (int t = wv; t < 64; t += 8) {
    const int mt = t & 3, nt = t >> 2;
    const float bias = b1[nt * 16 + n16];
    v8f acc;
#pragma unroll
    for (int j = 0; j < 8; ++j) acc[j] = bias;
#pragma unroll
    for (int kt = 0; kt < 12; ++kt) {
      v16h a = lds_a_frag(sX + mt * 16 * 384 + kt * 32, 384);
      v16h b = ldg_b_frag(w1s, 12, kt, nt);
      acc = __builtin_amdgcn_wmma_f32_16x16x32_f16(false, a, false, b,
                                                   (short)0, acc, false, false);
    }
#pragma unroll
    for (int j = 0; j < 8; ++j) {
      const int m = mt * 16 + hlf * 8 + j;               // D: row = vgpr + 8*half
      sH[m * 256 + nt * 16 + n16] = (_Float16)silu_f(acc[j]);
    }
  }
  __syncthreads();

  // Stage C: O = H @ W2 + Xe @ Wr + (b2 + br), 64x128
  for (int t = wv; t < 32; t += 8) {
    const int mt = t & 3, nt = t >> 2;
    const float bias = b2[nt * 16 + n16] + br[nt * 16 + n16];
    v8f acc;
#pragma unroll
    for (int j = 0; j < 8; ++j) acc[j] = bias;
#pragma unroll
    for (int kt = 0; kt < 8; ++kt) {                     // K = 256
      v16h a = lds_a_frag(sH + mt * 16 * 256 + kt * 32, 256);
      v16h b = ldg_b_frag(w2s, 8, kt, nt);
      acc = __builtin_amdgcn_wmma_f32_16x16x32_f16(false, a, false, b,
                                                   (short)0, acc, false, false);
    }
#pragma unroll
    for (int kt = 0; kt < 12; ++kt) {                    // K = 384 residual path
      v16h a = lds_a_frag(sX + mt * 16 * 384 + kt * 32, 384);
      v16h b = ldg_b_frag(wrs, 12, kt, nt);
      acc = __builtin_amdgcn_wmma_f32_16x16x32_f16(false, a, false, b,
                                                   (short)0, acc, false, false);
    }
#pragma unroll
    for (int j = 0; j < 8; ++j) {
      const int m = mt * 16 + hlf * 8 + j;
      sO[m * 132 + nt * 16 + n16] = acc[j];
    }
  }
  __syncthreads();

  // Stage D: per-row LayerNorm (quad of lanes per row), write + scatter-add
  const int row = tid >> 2, sub = tid & 3;
  const float* rp = sO + row * 132 + sub * 32;
  float s = 0.0f, ss = 0.0f;
#pragma unroll
  for (int c = 0; c < 32; ++c) { const float v = rp[c]; s += v; ss += v * v; }
  s  += __shfl_xor(s, 1, 32);  s  += __shfl_xor(s, 2, 32);
  ss += __shfl_xor(ss, 1, 32); ss += __shfl_xor(ss, 2, 32);
  const float mean = s * (1.0f / 128.0f);
  const float var = ss * (1.0f / 128.0f) - mean * mean;
  const float rstd = rsqrtf(var + 1e-5f);
  const int d = (int)eidx[EE + e0 + row];
  float* op = out_edge + (size_t)(e0 + row) * ED + sub * 32;
  float* ap = agg + (size_t)d * ED + sub * 32;
#pragma unroll
  for (int c = 0; c < 32; ++c) {
    const int cc = sub * 32 + c;
    const float y = (rp[c] - mean) * rstd * gamma[cc] + beta[cc];
    op[c] = y;
    atomicAdd(ap + c, y);
  }
}

// ---------------------------------------------------------------------------
// Fused node-update kernel: Xn=[64,256] concat(x, agg) -> MLP+res -> LN -> new_x
// ---------------------------------------------------------------------------
__global__ void __launch_bounds__(256)
node_kernel(const float* __restrict__ x, const float* __restrict__ agg,
            const _Float16* __restrict__ w1s, const float* __restrict__ b1,
            const _Float16* __restrict__ w2s, const float* __restrict__ b2,
            const _Float16* __restrict__ wrs, const float* __restrict__ br,
            const float* __restrict__ gamma, const float* __restrict__ beta,
            float* __restrict__ out_x) {
  extern __shared__ char smem[];
  _Float16* sX = (_Float16*)smem;                              // 64 x 256 f16
  _Float16* sH = (_Float16*)(smem + 64 * 256 * 2);             // 64 x 256 f16
  float*    sO = (float*)(smem + 2 * 64 * 256 * 2);            // 64 x 132 f32

  const int tid = threadIdx.x;
  const int wv = tid >> 5;
  const int lane = tid & 31;
  const int n16 = lane & 15, hlf = lane >> 4;
  const int n0 = blockIdx.x * 64;

  // Stage A: [x | agg] -> f16 LDS (zero-pad tail rows)
  for (int i = tid; i < 64 * 256; i += 256) {
    const int r = i >> 8, c = i & 255;
    const int node = n0 + r;
    float v = 0.0f;
    if (node < NN)
      v = (c < ND) ? x[(size_t)node * ND + c] : agg[(size_t)node * ED + (c - ND)];
    sX[i] = (_Float16)v;
  }
  __syncthreads();

  // Stage B: H = silu(Xn @ nw1 + nb1), 64x256, K=256
  for (int t = wv; t < 64; t += 8) {
    const int mt = t & 3, nt = t >> 2;
    const float bias = b1[nt * 16 + n16];
    v8f acc;
#pragma unroll
    for (int j = 0; j < 8; ++j) acc[j] = bias;
#pragma unroll
    for (int kt = 0; kt < 8; ++kt) {
      v16h a = lds_a_frag(sX + mt * 16 * 256 + kt * 32, 256);
      v16h b = ldg_b_frag(w1s, 8, kt, nt);
      acc = __builtin_amdgcn_wmma_f32_16x16x32_f16(false, a, false, b,
                                                   (short)0, acc, false, false);
    }
#pragma unroll
    for (int j = 0; j < 8; ++j) {
      const int m = mt * 16 + hlf * 8 + j;
      sH[m * 256 + nt * 16 + n16] = (_Float16)silu_f(acc[j]);
    }
  }
  __syncthreads();

  // Stage C: O = H @ nw2 + Xn @ nwr + (nb2 + nbr), 64x128, both K=256
  for (int t = wv; t < 32; t += 8) {
    const int mt = t & 3, nt = t >> 2;
    const float bias = b2[nt * 16 + n16] + br[nt * 16 + n16];
    v8f acc;
#pragma unroll
    for (int j = 0; j < 8; ++j) acc[j] = bias;
#pragma unroll
    for (int kt = 0; kt < 8; ++kt) {
      v16h a = lds_a_frag(sH + mt * 16 * 256 + kt * 32, 256);
      v16h b = ldg_b_frag(w2s, 8, kt, nt);
      acc = __builtin_amdgcn_wmma_f32_16x16x32_f16(false, a, false, b,
                                                   (short)0, acc, false, false);
    }
#pragma unroll
    for (int kt = 0; kt < 8; ++kt) {
      v16h a = lds_a_frag(sX + mt * 16 * 256 + kt * 32, 256);
      v16h b = ldg_b_frag(wrs, 8, kt, nt);
      acc = __builtin_amdgcn_wmma_f32_16x16x32_f16(false, a, false, b,
                                                   (short)0, acc, false, false);
    }
#pragma unroll
    for (int j = 0; j < 8; ++j) {
      const int m = mt * 16 + hlf * 8 + j;
      sO[m * 132 + nt * 16 + n16] = acc[j];
    }
  }
  __syncthreads();

  // Stage D: LayerNorm + guarded write
  const int row = tid >> 2, sub = tid & 3;
  const float* rp = sO + row * 132 + sub * 32;
  float s = 0.0f, ss = 0.0f;
#pragma unroll
  for (int c = 0; c < 32; ++c) { const float v = rp[c]; s += v; ss += v * v; }
  s  += __shfl_xor(s, 1, 32);  s  += __shfl_xor(s, 2, 32);
  ss += __shfl_xor(ss, 1, 32); ss += __shfl_xor(ss, 2, 32);
  const float mean = s * (1.0f / 128.0f);
  const float var = ss * (1.0f / 128.0f) - mean * mean;
  const float rstd = rsqrtf(var + 1e-5f);
  if (n0 + row < NN) {
    float* op = out_x + (size_t)(n0 + row) * ND + sub * 32;
#pragma unroll
    for (int c = 0; c < 32; ++c) {
      const int cc = sub * 32 + c;
      op[c] = (rp[c] - mean) * rstd * gamma[cc] + beta[cc];
    }
  }
}

// ---------------------------------------------------------------------------
// Launch
// ---------------------------------------------------------------------------
extern "C" void kernel_launch(void* const* d_in, const int* in_sizes, int n_in,
                              void* d_out, int out_size, void* d_ws, size_t ws_size,
                              hipStream_t stream) {
  const float* x        = (const float*)d_in[0];
  const float* eattr    = (const float*)d_in[1];
  const long long* eidx = (const long long*)d_in[2];  // int64 per reference
  const float* ew1 = (const float*)d_in[3];  const float* eb1 = (const float*)d_in[4];
  const float* ew2 = (const float*)d_in[5];  const float* eb2 = (const float*)d_in[6];
  const float* ewr = (const float*)d_in[7];  const float* ebr = (const float*)d_in[8];
  const float* egm = (const float*)d_in[9];  const float* ebt = (const float*)d_in[10];
  const float* nw1 = (const float*)d_in[11]; const float* nb1 = (const float*)d_in[12];
  const float* nw2 = (const float*)d_in[13]; const float* nb2 = (const float*)d_in[14];
  const float* nwr = (const float*)d_in[15]; const float* nbr = (const float*)d_in[16];
  const float* ngm = (const float*)d_in[17]; const float* nbt = (const float*)d_in[18];

  // workspace layout: agg [N,ED] f32, then f16-swizzled weights
  float* agg = (float*)d_ws;
  _Float16* wb = (_Float16*)((char*)d_ws + (size_t)NN * ED * sizeof(float));
  _Float16* ew1s = wb;
  _Float16* ew2s = ew1s + 384 * 256;
  _Float16* ewrs = ew2s + 256 * 128;
  _Float16* nw1s = ewrs + 384 * 128;
  _Float16* nw2s = nw1s + 256 * 256;
  _Float16* nwrs = nw2s + 256 * 128;

  float* out_x = (float*)d_out;                 // [N, ND]
  float* out_e = out_x + (size_t)NN * ND;       // [E, ED]

  // deterministic per-call prep: zero agg, swizzle weights to f16
  zero_f32_kernel<<<(NN * ED + 255) / 256, 256, 0, stream>>>(agg, NN * ED);
  swizzle_w_kernel<<<(12 * 16 * 32 + 255) / 256, 256, 0, stream>>>(ew1, ew1s, 384, 256);
  swizzle_w_kernel<<<( 8 *  8 * 32 + 255) / 256, 256, 0, stream>>>(ew2, ew2s, 256, 128);
  swizzle_w_kernel<<<(12 *  8 * 32 + 255) / 256, 256, 0, stream>>>(ewr, ewrs, 384, 128);
  swizzle_w_kernel<<<( 8 * 16 * 32 + 255) / 256, 256, 0, stream>>>(nw1, nw1s, 256, 256);
  swizzle_w_kernel<<<( 8 *  8 * 32 + 255) / 256, 256, 0, stream>>>(nw2, nw2s, 256, 128);
  swizzle_w_kernel<<<( 8 *  8 * 32 + 255) / 256, 256, 0, stream>>>(nwr, nwrs, 256, 128);

  const size_t edge_smem = 64 * 384 * 2 + 64 * 256 * 2 + 64 * 132 * 4;  // ~113 KB
  const size_t node_smem = 2 * (64 * 256 * 2) + 64 * 132 * 4;           // ~97 KB

  edge_kernel<<<EE / 64, 256, edge_smem, stream>>>(
      x, eattr, eidx, ew1s, eb1, ew2s, eb2, ewrs, ebr, egm, ebt, out_e, agg);
  node_kernel<<<(NN + 63) / 64, 256, node_smem, stream>>>(
      x, agg, nw1s, nb1, nw2s, nb2, nwrs, nbr, ngm, nbt, out_x);
}
